// TransformerOCModel_22162031247522
// MI455X (gfx1250) — matrix operand order repaired and checked by reading, not compiled, not run on hardware
//
#include <hip/hip_runtime.h>
#include <hip/hip_bf16.h>
#include <math.h>

typedef unsigned short u16;
typedef unsigned int   u32;
typedef __attribute__((ext_vector_type(16))) __bf16 v16bf;
typedef __attribute__((ext_vector_type(8)))  float  v8f;

#define NPTS   4096
#define DIN    49
#define HID    256
#define NLAYER 4
#define NHEAD  8
#define DHEAD  32
#define LATD   16
#define FFD    512

__constant__ float c_min[49] = {
  -3.4627f,-2.4507f,-3.1416f,-0.2362f,-0.7272f,-0.8202f,-0.8768f,-0.5988f,-0.2704f,-0.8894f,
  -1.059f,-1.3935f,-0.7862f,-73.8012f,-72.4212f,-187.575f,-73.8019f,-77.1076f,-187.8705f,-0.0231f,
  -0.5393f,-5.025f,-93.8255f,-93.9099f,-223.854f,-97.9344f,-93.906f,-224.1495f,-0.0111f,-0.8231f,
  -5.0255f,-93.8255f,-93.9099f,-223.854f,-97.9344f,-93.906f,-224.1495f,-0.0111f,-0.8231f,-5.0255f,
  -110.016f,-110.015f,-267.235f,-110.1943f,-110.195f,-267.635f,-0.0046f,-0.9416f,-5.0259f };
__constant__ float c_max[49] = {
  7.5964f,2.4535f,3.1416f,0.2434f,0.7304f,0.7744f,0.8728f,0.5682f,0.2755f,0.8922f,
  1.15f,1.5453f,0.8093f,73.7524f,70.415f,187.575f,73.7676f,71.4665f,187.8705f,0.0228f,
  0.5575f,5.025f,93.8869f,93.8406f,223.854f,96.144f,96.9953f,224.1495f,0.0097f,0.8098f,
  5.0253f,93.8869f,93.8406f,223.854f,96.144f,96.9953f,224.1495f,0.0097f,0.8098f,5.0253f,
  110.0128f,110.015f,267.235f,110.1814f,110.195f,267.635f,0.005f,0.9323f,5.026f };

__device__ __forceinline__ u16 f2bf(float f){
  u32 x = __float_as_uint(f);
  u32 r = x + 0x7fffu + ((x >> 16) & 1u);
  return (u16)(r >> 16);
}
__device__ __forceinline__ float bf2f(u16 u){ return __uint_as_float(((u32)u) << 16); }

union FragU { v16bf v; uint4 q[2]; };

// A-matrix 16x32 bf16 fragment: per lane (m = lane&15, half = lane>>4)
// elems 0..15 -> k = 16*half + [0,16), contiguous
__device__ __forceinline__ v16bf load_a_frag(const u16* rowp, int hf){
  FragU u;
  u.q[0] = *(const uint4*)(rowp + 16*hf);
  u.q[1] = *(const uint4*)(rowp + 16*hf + 8);
  return u.v;
}
// B-matrix 32x16 bf16 fragment: per lane (n = lane&15, half = lane>>4)
// elems 0..15 -> k = 16*half + [0,16), contiguous; caller passes p = base + 16*half
__device__ __forceinline__ v16bf load_b_frag(const u16* p){
  FragU u;
  u.q[0] = *(const uint4*)(p);
  u.q[1] = *(const uint4*)(p + 8);
  return u.v;
}

__device__ __forceinline__ float mask_val(float qe, float qp, float ke, float kp){
  float de = qe - ke;
  float dp = qp - kp;
  dp -= 6.28318530718f * rintf(dp * 0.15915494309f);   // wrap to (-pi, pi]
  float dr2 = de*de + dp*dp;
  return (dr2 <= 0.04000001f) ? 0.f : -3.0e38f;
}

// ---------------- elementwise / prep kernels ----------------

__global__ void f32_to_bf16_kernel(const float* __restrict__ src, u16* __restrict__ dst, int n){
  int i = blockIdx.x*256 + threadIdx.x;
  if (i < n) dst[i] = f2bf(src[i]);
}

__global__ void etaphi_kernel(const float* __restrict__ xr, float* __restrict__ ep, int n){
  int i = blockIdx.x*blockDim.x + threadIdx.x;
  if (i >= n) return;
  const float* row = xr + (size_t)i*DIN;
#define DN(c) (row[(c)]*(c_max[(c)]-c_min[(c)]+1e-8f)+c_min[(c)])
  float pt  = fmaxf(__expf(DN(0)), 0.5f);
  float eta = DN(1);
  float phi = DN(2);
  float r = 0.f;
  const int offs[4] = {13,22,31,40};
#pragma unroll
  for (int k=0;k<4;k++){ float a=DN(offs[k]), b=DN(offs[k]+1); r += sqrtf(a*a+b*b); }
  r *= 0.25f;
  float rh  = pt / (0.3f*3.8f);
  float arg = (3600.f - r*r) / (2.f*r*rh + 1e-6f);
  arg = fminf(fmaxf(arg, -1.f), 1.f);
  ep[2*i]   = eta;
  ep[2*i+1] = phi + asinf(arg);
#undef DN
}

__global__ void proj_kernel(const float* __restrict__ xr, const float* __restrict__ Wp,
                            const float* __restrict__ bp, float* __restrict__ x){
  int n = blockIdx.x, h = threadIdx.x;
  float acc = bp[h];
#pragma unroll 7
  for (int d=0; d<DIN; d++) acc += xr[(size_t)n*DIN + d] * Wp[(size_t)h*DIN + d];
  x[(size_t)n*HID + h] = acc;
}

__global__ void ln_bf16_kernel(const float* __restrict__ x, const float* __restrict__ w,
                               const float* __restrict__ b, u16* __restrict__ out){
  __shared__ float red[HID];
  int row = blockIdx.x, t = threadIdx.x;
  float v = x[(size_t)row*HID + t];
  red[t] = v; __syncthreads();
  for (int s=HID/2; s>0; s>>=1){ if (t<s) red[t]+=red[t+s]; __syncthreads(); }
  float mean = red[0] * (1.f/HID);
  __syncthreads();
  float d = v - mean;
  red[t] = d*d; __syncthreads();
  for (int s=HID/2; s>0; s>>=1){ if (t<s) red[t]+=red[t+s]; __syncthreads(); }
  float var = red[0] * (1.f/HID);
  float y = d * rsqrtf(var + 1e-5f) * w[t] + b[t];
  out[(size_t)row*HID + t] = f2bf(y);
}

// qkv[N][768] -> vt[(h*32+d)][N]
__global__ void vtrans_kernel(const u16* __restrict__ qkv, u16* __restrict__ vt){
  int idx = blockIdx.x*256 + threadIdx.x;
  if (idx >= NPTS*HID) return;
  int n = idx >> 8, c = idx & 255;
  vt[(size_t)c*NPTS + n] = qkv[(size_t)n*(3*HID) + 2*HID + c];
}

__global__ void coords_kernel(const float* __restrict__ cp, float* __restrict__ out, int n){
  int i = blockIdx.x*256 + threadIdx.x;
  if (i >= n) return;
  float s = 0.f;
#pragma unroll
  for (int j=0;j<LATD;j++){ float v = cp[(size_t)i*LATD+j]; s += v*v; }
  float inv = 1.f / fmaxf(sqrtf(s), 1e-12f);
#pragma unroll
  for (int j=0;j<LATD;j++) out[(size_t)i*LATD + j] = cp[(size_t)i*LATD+j]*inv;
}

__global__ void beta_kernel(const u16* __restrict__ hb, const float* __restrict__ bw2,
                            const float* __restrict__ bb2, float* __restrict__ out, int n){
  int i = blockIdx.x*256 + threadIdx.x;
  if (i >= n) return;
  float acc = bb2[0];
#pragma unroll 8
  for (int j=0;j<128;j++) acc += bf2f(hb[(size_t)i*128 + j]) * bw2[j];
  float b = 1.f / (1.f + __expf(-acc));
  out[i] = fminf(fmaxf(b, 1e-6f), 1.f - 1e-6f);
}

// ---------------- WMMA GEMM: C[m][n] = sum_k A[m][k]*W[n][k] (+bias, relu, residual) ----------------
// block = 256 thr (8 waves); block tile 128(M) x 16*NT(N); wave tile 16 x 16*NT; K step 32.
// Branch-free inner loop: Nout must be a multiple of 16*NT (no tile guards -> no acc reg copies).
template<int NT>
__global__ void __launch_bounds__(256)
gemm_bf16_kernel(const u16* __restrict__ A, int lda,
                 const u16* __restrict__ W, int ldw,
                 const float* __restrict__ bias,
                 const float* __restrict__ residual, int ldr,
                 float* __restrict__ Cf, int ldcf,
                 u16* __restrict__ Cb, int ldcb,
                 int K, int relu)
{
  const int wv = threadIdx.x >> 5, lane = threadIdx.x & 31;
  const int hf = lane >> 4, ln = lane & 15;
  const int m0 = blockIdx.x*128 + wv*16;
  const int n0 = blockIdx.y*16*NT;
  const u16* arow = A + (size_t)(m0 + ln) * lda;  // A-frag row (same for both lane halves)

  v8f acc[NT] = {};
  const u16* wrow[NT];
#pragma unroll
  for (int t=0;t<NT;t++)
    wrow[t] = W + (size_t)(n0 + 16*t + ln) * ldw + 16*hf;

#pragma unroll 4
  for (int k0=0; k0<K; k0+=32){
    v16bf a = load_a_frag(arow + k0, hf);
#pragma unroll
    for (int t=0;t<NT;t++){
      v16bf b = load_b_frag(wrow[t] + k0);
      acc[t] = __builtin_amdgcn_wmma_f32_16x16x32_bf16(false, a, false, b,
                                                       (short)0, acc[t], false, false);
    }
  }

  // epilogue: uniform feature branches hoisted out of per-row loops
  float v[NT][8];
#pragma unroll
  for (int t=0;t<NT;t++){
    float bs = bias[n0 + 16*t + ln];
#pragma unroll
    for (int r=0;r<8;r++){
      float x = acc[t][r] + bs;
      v[t][r] = relu ? fmaxf(x, 0.f) : x;
    }
  }
  if (residual){
#pragma unroll
    for (int t=0;t<NT;t++){
      int col = n0 + 16*t + ln;
#pragma unroll
      for (int r=0;r<8;r++)
        v[t][r] += residual[(size_t)(m0 + r + 8*hf)*ldr + col];
    }
  }
  if (Cf){
#pragma unroll
    for (int t=0;t<NT;t++){
      int col = n0 + 16*t + ln;
#pragma unroll
      for (int r=0;r<8;r++)
        Cf[(size_t)(m0 + r + 8*hf)*ldcf + col] = v[t][r];
    }
  }
  if (Cb){
#pragma unroll
    for (int t=0;t<NT;t++){
      int col = n0 + 16*t + ln;
#pragma unroll
      for (int r=0;r<8;r++)
        Cb[(size_t)(m0 + r + 8*hf)*ldcb + col] = f2bf(v[t][r]);
    }
  }
}

// ---------------- fused flash attention (per head, online softmax, dr-mask on the fly) -------------
// grid = (NPTS/128, NHEAD); 8 waves/block; each wave owns 16 query rows.
__global__ void __launch_bounds__(256)
attn_kernel(const u16* __restrict__ qkv, const u16* __restrict__ vt,
            const float* __restrict__ etaphi, u16* __restrict__ attnout)
{
  __shared__ __align__(16) u16 pbuf[8][16*32];   // per-wave P staging (C-layout -> A-layout)
  const int wv = threadIdx.x >> 5, lane = threadIdx.x & 31;
  const int hf = lane >> 4, ln = lane & 15;
  const int head = blockIdx.y;
  const int q0 = blockIdx.x*128 + wv*16;

  const u16* qrow = qkv + (size_t)(q0 + ln)*(3*HID) + head*DHEAD;
  v16bf qa = load_a_frag(qrow, hf);              // Q tile 16x32 (A-matrix)

  float qe[8], qp[8];
#pragma unroll
  for (int r=0;r<8;r++){
    int m = q0 + r + 8*hf;
    qe[r] = etaphi[2*m]; qp[r] = etaphi[2*m+1];
  }

  v8f o0 = {}, o1 = {};
  float mrow[8], lrow[8];
#pragma unroll
  for (int r=0;r<8;r++){ mrow[r] = -3.0e38f; lrow[r] = 0.f; }
  const float scale = 0.17677669529663687f;      // 1/sqrt(32)

  for (int kt=0; kt<NPTS; kt+=32){
    // K tiles as B-matrix (B[kk=dim][n=key]); contiguous head-dim runs per key row
    const u16* kp0 = qkv + (size_t)(kt      + ln)*(3*HID) + HID + head*DHEAD + 16*hf;
    const u16* kp1 = qkv + (size_t)(kt + 16 + ln)*(3*HID) + HID + head*DHEAD + 16*hf;
    v16bf kb0 = load_b_frag(kp0);
    v16bf kb1 = load_b_frag(kp1);
    v8f z = {};
    v8f s0 = __builtin_amdgcn_wmma_f32_16x16x32_bf16(false, qa, false, kb0, (short)0, z, false, false);
    v8f s1 = __builtin_amdgcn_wmma_f32_16x16x32_bf16(false, qa, false, kb1, (short)0, z, false, false);

    float ke0 = etaphi[2*(kt+ln)],    kh0 = etaphi[2*(kt+ln)+1];
    float ke1 = etaphi[2*(kt+16+ln)], kh1 = etaphi[2*(kt+16+ln)+1];

    float p0[8], p1[8];
#pragma unroll
    for (int r=0;r<8;r++){
      float a0 = s0[r]*scale + mask_val(qe[r], qp[r], ke0, kh0);
      float a1 = s1[r]*scale + mask_val(qe[r], qp[r], ke1, kh1);
      float mx = fmaxf(a0, a1);
#pragma unroll
      for (int d=1; d<16; d<<=1) mx = fmaxf(mx, __shfl_xor(mx, d, 32));
      float mnew = fmaxf(mrow[r], mx);
      float corr = __expf(mrow[r] - mnew);
      float e0 = __expf(a0 - mnew);
      float e1 = __expf(a1 - mnew);
      float rs = e0 + e1;
#pragma unroll
      for (int d=1; d<16; d<<=1) rs += __shfl_xor(rs, d, 32);
      lrow[r] = lrow[r]*corr + rs;
      mrow[r] = mnew;
      o0[r] *= corr; o1[r] *= corr;
      p0[r] = e0; p1[r] = e1;
    }
    // C-layout -> LDS row-major [16 q][32 key] -> reload as A-fragment (wave-private, in-order DS)
    u16* pb = pbuf[wv];
#pragma unroll
    for (int r=0;r<8;r++){
      pb[(r + 8*hf)*32 + ln]      = f2bf(p0[r]);
      pb[(r + 8*hf)*32 + 16 + ln] = f2bf(p1[r]);
    }
    v16bf pa = load_a_frag(pb + ln*32, hf);
    // V tiles as B-matrix from pre-transposed Vt[(h*32+d)][N]
    const u16* vp0 = vt + (size_t)(head*DHEAD + ln)*NPTS      + kt + 16*hf;
    const u16* vp1 = vt + (size_t)(head*DHEAD + 16 + ln)*NPTS + kt + 16*hf;
    v16bf vb0 = load_b_frag(vp0);
    v16bf vb1 = load_b_frag(vp1);
    o0 = __builtin_amdgcn_wmma_f32_16x16x32_bf16(false, pa, false, vb0, (short)0, o0, false, false);
    o1 = __builtin_amdgcn_wmma_f32_16x16x32_bf16(false, pa, false, vb1, (short)0, o1, false, false);
  }
#pragma unroll
  for (int r=0;r<8;r++){
    int m = q0 + r + 8*hf;
    float inv = 1.f / lrow[r];
    attnout[(size_t)m*HID + head*DHEAD + ln]      = f2bf(o0[r]*inv);
    attnout[(size_t)m*HID + head*DHEAD + 16 + ln] = f2bf(o1[r]*inv);
  }
}

// ---------------------------------------------------------------------------------------------------

extern "C" void kernel_launch(void* const* d_in, const int* in_sizes, int n_in,
                              void* d_out, int out_size, void* d_ws, size_t ws_size,
                              hipStream_t stream)
{
  (void)in_sizes; (void)n_in; (void)out_size; (void)ws_size;
  const float* x_raw = (const float*)d_in[0];
  // d_in[1] = batch (unused)
  const float* Wp    = (const float*)d_in[2];
  const float* bp    = (const float*)d_in[3];
  const float* ln1_w = (const float*)d_in[4];
  const float* ln1_b = (const float*)d_in[5];
  const float* qkv_w = (const float*)d_in[6];
  const float* qkv_b = (const float*)d_in[7];
  const float* out_w = (const float*)d_in[8];
  const float* out_b = (const float*)d_in[9];
  const float* ln2_w = (const float*)d_in[10];
  const float* ln2_b = (const float*)d_in[11];
  const float* ff1_w = (const float*)d_in[12];
  const float* ff1_b = (const float*)d_in[13];
  const float* ff2_w = (const float*)d_in[14];
  const float* ff2_b = (const float*)d_in[15];
  const float* lw1   = (const float*)d_in[16];
  const float* lb1   = (const float*)d_in[17];
  const float* lw2   = (const float*)d_in[18];
  const float* lb2   = (const float*)d_in[19];
  const float* bw1   = (const float*)d_in[20];
  const float* bb1   = (const float*)d_in[21];
  const float* bw2   = (const float*)d_in[22];
  const float* bb2   = (const float*)d_in[23];
  float* out = (float*)d_out;

  char* p = (char*)d_ws;
  auto alloc = [&](size_t bytes)->char*{ char* r = p; p += (bytes + 255) & ~(size_t)255; return r; };
  float* etaphi = (float*)alloc((size_t)NPTS*2*4);
  float* x      = (float*)alloc((size_t)NPTS*HID*4);
  u16*   hbf    = (u16*)  alloc((size_t)NPTS*HID*2);
  u16*   qkvbf  = (u16*)  alloc((size_t)NPTS*3*HID*2);
  u16*   vtbf   = (u16*)  alloc((size_t)HID*NPTS*2);
  u16*   aobf   = (u16*)  alloc((size_t)NPTS*HID*2);
  u16*   ffbf   = (u16*)  alloc((size_t)NPTS*FFD*2);
  u16*   xbf    = (u16*)  alloc((size_t)NPTS*HID*2);
  u16*   hlatbf = (u16*)  alloc((size_t)NPTS*HID*2);
  u16*   hbbf   = (u16*)  alloc((size_t)NPTS*128*2);
  float* cpre   = (float*)alloc((size_t)NPTS*LATD*4);
  u16* qkvw_bf  = (u16*)alloc((size_t)NLAYER*3*HID*HID*2);
  u16* outw_bf  = (u16*)alloc((size_t)NLAYER*HID*HID*2);
  u16* ff1w_bf  = (u16*)alloc((size_t)NLAYER*FFD*HID*2);
  u16* ff2w_bf  = (u16*)alloc((size_t)NLAYER*HID*FFD*2);
  u16* lw1_bf   = (u16*)alloc((size_t)HID*HID*2);
  u16* lw2_bf   = (u16*)alloc((size_t)LATD*HID*2);
  u16* bw1_bf   = (u16*)alloc((size_t)128*HID*2);

  auto conv = [&](const float* s, u16* d, int n){
    f32_to_bf16_kernel<<<(n+255)/256, 256, 0, stream>>>(s, d, n);
  };
  conv(qkv_w, qkvw_bf, NLAYER*3*HID*HID);
  conv(out_w, outw_bf, NLAYER*HID*HID);
  conv(ff1_w, ff1w_bf, NLAYER*FFD*HID);
  conv(ff2_w, ff2w_bf, NLAYER*HID*FFD);
  conv(lw1, lw1_bf, HID*HID);
  conv(lw2, lw2_bf, LATD*HID);
  conv(bw1, bw1_bf, 128*HID);

  etaphi_kernel<<<(NPTS+255)/256, 256, 0, stream>>>(x_raw, etaphi, NPTS);
  proj_kernel<<<NPTS, HID, 0, stream>>>(x_raw, Wp, bp, x);

  // Nout % 64 == 0 -> NT=4; else (latent head, Nout=16) -> NT=1
  auto gemm = [&](const u16* A, int lda, const u16* W, int ldw, const float* bias,
                  const float* res, int ldr, float* Cf, int ldcf, u16* Cb, int ldcb,
                  int M, int Nout, int K, int relu){
    if ((Nout & 63) == 0){
      dim3 g(M/128, Nout/64);
      gemm_bf16_kernel<4><<<g, 256, 0, stream>>>(A, lda, W, ldw, bias, res, ldr,
                                                 Cf, ldcf, Cb, ldcb, K, relu);
    } else {
      dim3 g(M/128, Nout/16);
      gemm_bf16_kernel<1><<<g, 256, 0, stream>>>(A, lda, W, ldw, bias, res, ldr,
                                                 Cf, ldcf, Cb, ldcb, K, relu);
    }
  };

  for (int l=0; l<NLAYER; l++){
    ln_bf16_kernel<<<NPTS, HID, 0, stream>>>(x, ln1_w + (size_t)l*HID, ln1_b + (size_t)l*HID, hbf);
    gemm(hbf, HID, qkvw_bf + (size_t)l*3*HID*HID, HID, qkv_b + (size_t)l*3*HID,
         nullptr, 0, nullptr, 0, qkvbf, 3*HID, NPTS, 3*HID, HID, 0);
    vtrans_kernel<<<(NPTS*HID+255)/256, 256, 0, stream>>>(qkvbf, vtbf);
    dim3 ag(NPTS/128, NHEAD);
    attn_kernel<<<ag, 256, 0, stream>>>(qkvbf, vtbf, etaphi, aobf);
    gemm(aobf, HID, outw_bf + (size_t)l*HID*HID, HID, out_b + (size_t)l*HID,
         x, HID, x, HID, nullptr, 0, NPTS, HID, HID, 0);
    ln_bf16_kernel<<<NPTS, HID, 0, stream>>>(x, ln2_w + (size_t)l*HID, ln2_b + (size_t)l*HID, hbf);
    gemm(hbf, HID, ff1w_bf + (size_t)l*FFD*HID, HID, ff1_b + (size_t)l*FFD,
         nullptr, 0, nullptr, 0, ffbf, FFD, NPTS, FFD, HID, 1);
    gemm(ffbf, FFD, ff2w_bf + (size_t)l*HID*FFD, FFD, ff2_b + (size_t)l*HID,
         x, HID, x, HID, nullptr, 0, NPTS, HID, FFD, 0);
  }

  conv(x, xbf, NPTS*HID);
  gemm(xbf, HID, lw1_bf, HID, lb1, nullptr, 0, nullptr, 0, hlatbf, HID, NPTS, HID, HID, 1);
  gemm(hlatbf, HID, lw2_bf, HID, lb2, nullptr, 0, cpre, LATD, nullptr, 0, NPTS, LATD, HID, 0);
  coords_kernel<<<(NPTS+255)/256, 256, 0, stream>>>(cpre, out + NPTS, NPTS);
  gemm(xbf, HID, bw1_bf, HID, bb1, nullptr, 0, nullptr, 0, hbbf, 128, NPTS, 128, HID, 1);
  beta_kernel<<<(NPTS+255)/256, 256, 0, stream>>>(hbbf, bw2, bb2, out, NPTS);
}
